// NonLocalBlock_5600637354143
// MI455X (gfx1250) — matrix operand order repaired and checked by reading, not compiled, not run on hardware
//
#include <hip/hip_runtime.h>
#include <hip/hip_bf16.h>
#include <cstdint>
#include <cstddef>

#define BVAL  8
#define CVAL  256
#define DIVAL 128
#define DOVAL 256
#define NVAL  3136   // 56*56, = 196*16

typedef __attribute__((ext_vector_type(16))) __bf16 bf16x16;
typedef __attribute__((ext_vector_type(8)))  float  f32x8;
typedef unsigned int u32x4 __attribute__((ext_vector_type(4)));
typedef int          i32x8 __attribute__((ext_vector_type(8)));
typedef int          i32x4 __attribute__((ext_vector_type(4)));

#if defined(__has_builtin)
#if __has_builtin(__builtin_amdgcn_tensor_load_to_lds) && \
    __has_builtin(__builtin_amdgcn_s_wait_tensorcnt)
#define USE_TDM 1
#endif
#endif

union Frag {
  bf16x16 v;
  uint4 q[2];
  unsigned short u[16];
};

__device__ __forceinline__ f32x8 zero8() {
  f32x8 z = {0.f, 0.f, 0.f, 0.f, 0.f, 0.f, 0.f, 0.f};
  return z;
}

// Native f32 -> bf16 conversion (backend picks the HW convert when available).
__device__ __forceinline__ unsigned short f2bf(float f) {
  __bf16 h = (__bf16)f;
  unsigned short b;
  __builtin_memcpy(&b, &h, 2);
  return b;
}

// A fragment (16xK row slice, bf16): per-lane row base, 16 consecutive k at k0+half*16
__device__ __forceinline__ Frag load_a_bf16(const unsigned short* rowbase, int k0, int half) {
  Frag f;
  f.q[0] = *(const uint4*)(rowbase + k0 + half * 16);
  f.q[1] = *(const uint4*)(rowbase + k0 + half * 16 + 8);
  return f;
}
// B fragment (Kx16): per-lane column base (layout [col][k]), 16 consecutive k at k0+half*16
__device__ __forceinline__ Frag load_b_bf16(const unsigned short* colbase, int k0, int half) {
  Frag f;
  f.q[0] = *(const uint4*)(colbase + k0 + half * 16);
  f.q[1] = *(const uint4*)(colbase + k0 + half * 16 + 8);
  return f;
}

__device__ __forceinline__ f32x8 wmma_bf16(const Frag& a, const Frag& b, f32x8 c) {
  return __builtin_amdgcn_wmma_f32_16x16x32_bf16(false, a.v, false, b.v, (short)0, c,
                                                 false, false);
}

// -------------------------------------------------------------------------
// Kernel 0: one-shot f32 -> bf16 weight conversion into workspace.
// -------------------------------------------------------------------------
__global__ void __launch_bounds__(256)
cvt_kernel(const float* __restrict__ in, unsigned short* __restrict__ out, int count) {
  int i = blockIdx.x * 256 + threadIdx.x;
  if (i < count) out[i] = f2bf(in[i]);
}

// -------------------------------------------------------------------------
// Kernel 1: fused theta/phi/g 1x1-conv projections (GEMM [DI x C] * [C x N]).
// x tiles are pulled into LDS by the Tensor Data Mover (TENSORcnt path) when
// the toolchain exposes the builtin; otherwise staged via VGPRs.
// theta/phi stored n-major [B][N][DI] (bf16), g stored d-major [B][DI][N].
// -------------------------------------------------------------------------
__global__ void __launch_bounds__(128)
proj_kernel(const float* __restrict__ x,
            const unsigned short* __restrict__ twb, const float* __restrict__ tb,
            const unsigned short* __restrict__ pwb, const float* __restrict__ pb,
            const unsigned short* __restrict__ gwb, const float* __restrict__ gb,
            unsigned short* __restrict__ theta_t,
            unsigned short* __restrict__ phi_t,
            unsigned short* __restrict__ g_dm)
{
  const int b     = blockIdx.y;
  const int nblk  = blockIdx.x % (NVAL / 64);
  const int diblk = blockIdx.x / (NVAL / 64);
  const int tid   = threadIdx.x;
  const int wave  = tid >> 5, lane = tid & 31;
  const int half  = lane >> 4, lr = lane & 15;
  const int di0   = diblk * 16;
  const int nbase = nblk * 64;

  extern __shared__ __align__(16) char smem_raw[];   // 8 KB dynamic LDS

  f32x8 accT = zero8(), accP = zero8(), accG = zero8();
  const float* xb = x + (size_t)b * CVAL * NVAL;
  const unsigned short* trow = twb + (size_t)(di0 + lr) * CVAL;
  const unsigned short* prow = pwb + (size_t)(di0 + lr) * CVAL;
  const unsigned short* grow = gwb + (size_t)(di0 + lr) * CVAL;

  for (int c0 = 0; c0 < CVAL; c0 += 32) {
    __syncthreads();
#if defined(USE_TDM)
    float* xsf = (float*)smem_raw;                   // [32 c][64 n] f32 tile
    if (wave == 0) {
      // Tensor DMA descriptor (ISA 8.3/8.4): 2D tile, f32, 32 rows x 64 cols,
      // row stride = NVAL elements; dynamic LDS starts at offset 0.
      unsigned long long ga =
          (unsigned long long)(uintptr_t)(xb + (size_t)c0 * NVAL + nbase);
      u32x4 g0 = { 1u,                                  // count=1, user desc
                   0u,                                  // lds_addr
                   (unsigned)ga,                        // global_addr[31:0]
                   (unsigned)((ga >> 32) & 0x1FFFFFFu)  // global_addr[56:32]
                       | (2u << 30) };                  // type = image
      i32x8 g1 = { (int)(2u << 16),          // data_size = 4 bytes
                   (int)((unsigned)NVAL << 16), // tensor_dim0[15:0]
                   (int)(256u << 16),        // tensor_dim0[31:16]=0 | tensor_dim1=256
                   (int)(64u << 16),         // tensor_dim1[31:16]=0 | tile_dim0=64
                   32,                       // tile_dim1=32 | tile_dim2=0
                   NVAL,                     // tensor_dim0_stride[31:0]
                   0, 0 };                   // stride hi, tensor_dim1_stride
      i32x4 gz4 = {0, 0, 0, 0};
      i32x8 gz8 = {0, 0, 0, 0, 0, 0, 0, 0};
      __builtin_amdgcn_tensor_load_to_lds(g0, g1, gz4, gz4, gz8, 0);
      __builtin_amdgcn_s_wait_tensorcnt(0);
    }
    __syncthreads();
    Frag bx;                                           // B[k=c][col=n]
#pragma unroll
    for (int e = 0; e < 16; ++e)
      bx.u[e] = f2bf(xsf[(half * 16 + e) * 64 + wave * 16 + lr]);
#else
    unsigned short* xs = (unsigned short*)smem_raw;    // [64 n][32 c] bf16
#pragma unroll
    for (int i = 0; i < 16; ++i) {
      int idx = i * 128 + tid;
      int cl = idx >> 6, nl = idx & 63;
      float v = xb[(size_t)(c0 + cl) * NVAL + nbase + nl];   // coalesced over n
      xs[nl * 32 + cl] = f2bf(v);                            // transpose into LDS
    }
    __syncthreads();
    Frag bx = load_b_bf16(xs + (wave * 16 + lr) * 32, 0, half);
#endif
    Frag aT = load_a_bf16(trow, c0, half);
    Frag aP = load_a_bf16(prow, c0, half);
    Frag aG = load_a_bf16(grow, c0, half);
    accT = wmma_bf16(aT, bx, accT);
    accP = wmma_bf16(aP, bx, accP);
    accG = wmma_bf16(aG, bx, accG);
  }

  const int n      = nbase + wave * 16 + lr;   // D-frag column
  const int dibase = di0 + half * 8;           // D-frag rows: dibase + r
  {
    Frag ft, fp;
#pragma unroll
    for (int r = 0; r < 8; ++r) {
      ft.u[r] = f2bf(accT[r] + tb[dibase + r]);
      fp.u[r] = f2bf(accP[r] + pb[dibase + r]);
    }
    *(uint4*)(theta_t + ((size_t)b * NVAL + n) * DIVAL + dibase) = ft.q[0];
    *(uint4*)(phi_t   + ((size_t)b * NVAL + n) * DIVAL + dibase) = fp.q[0];
  }
#pragma unroll
  for (int r = 0; r < 8; ++r) {
    int di = dibase + r;
    g_dm[((size_t)b * DIVAL + di) * NVAL + n] = f2bf(accG[r] + gb[di]);
  }
}

// -------------------------------------------------------------------------
// Kernel 2: fused flash attention. Each wave owns 16 query rows; stream over
// key/value columns in chunks of 32; scores never touch HBM.
// -------------------------------------------------------------------------
__global__ void __launch_bounds__(128)
attn_kernel(const unsigned short* __restrict__ theta_t,
            const unsigned short* __restrict__ phi_t,
            const unsigned short* __restrict__ g_dm,
            unsigned short* __restrict__ t_t)
{
  const int b    = blockIdx.y;
  const int tid  = threadIdx.x;
  const int wave = tid >> 5, lane = tid & 31;
  const int half = lane >> 4, lr = lane & 15;
  const int n    = blockIdx.x * 64 + wave * 16 + lr;

  __shared__ __align__(16) unsigned short p_lds[4][16 * 32]; // per-wave p tile
  __shared__ float red[4][16];                               // per-wave row->col bcast

  Frag aQ[4];
  const unsigned short* qrow = theta_t + ((size_t)b * NVAL + n) * DIVAL;
#pragma unroll
  for (int kc = 0; kc < 4; ++kc) aQ[kc] = load_a_bf16(qrow, kc * 32, half);

  f32x8 acc[8];
#pragma unroll
  for (int dt = 0; dt < 8; ++dt) acc[dt] = zero8();
  float mrun[8], lrun[8];
#pragma unroll
  for (int r = 0; r < 8; ++r) { mrun[r] = -3.0e38f; lrun[r] = 0.f; }

  const float sc = 0.08838834764831845f;  // DI^-0.5 = 1/sqrt(128)
  const unsigned short* phib  = phi_t + (size_t)b * NVAL * DIVAL;
  const unsigned short* gbase = g_dm + (size_t)b * DIVAL * NVAL;

  for (int m0 = 0; m0 < NVAL; m0 += 32) {
    // prefetch next m-tile's phi rows and g columns into cache
    if (m0 + 32 < NVAL) {
      __builtin_prefetch(phib + (size_t)(m0 + 32 + lr) * DIVAL, 0, 0);
      __builtin_prefetch(phib + (size_t)(m0 + 48 + lr) * DIVAL, 0, 0);
      __builtin_prefetch(gbase + (size_t)lr * NVAL + m0 + 32, 0, 0);
    }
    f32x8 s0 = zero8(), s1 = zero8();
#pragma unroll
    for (int kc = 0; kc < 4; ++kc) {
      Frag b0 = load_b_bf16(phib + (size_t)(m0 + lr) * DIVAL, kc * 32, half);
      Frag b1 = load_b_bf16(phib + (size_t)(m0 + 16 + lr) * DIVAL, kc * 32, half);
      s0 = wmma_bf16(aQ[kc], b0, s0);
      s1 = wmma_bf16(aQ[kc], b1, s1);
    }
    // online softmax: rows live across the 16 lanes of each half (row = r+8*half)
    float alpha[8];
#pragma unroll
    for (int r = 0; r < 8; ++r) {
      float v0 = s0[r] * sc, v1 = s1[r] * sc;
      float rm = fmaxf(v0, v1);
#pragma unroll
      for (int mk = 8; mk >= 1; mk >>= 1) rm = fmaxf(rm, __shfl_xor(rm, mk, 32));
      float mn = fmaxf(mrun[r], rm);
      alpha[r] = __expf(mrun[r] - mn);
      float p0 = __expf(v0 - mn);
      float p1 = __expf(v1 - mn);
      s0[r] = p0; s1[r] = p1;
      float rs = p0 + p1;
#pragma unroll
      for (int mk = 8; mk >= 1; mk >>= 1) rs += __shfl_xor(rs, mk, 32);
      lrun[r] = lrun[r] * alpha[r] + rs;
      mrun[r] = mn;
    }
    // stage p tile (16 rows x 32 cols) as bf16 for the B operand of the apply GEMM
#pragma unroll
    for (int r = 0; r < 8; ++r) {
      int row = r + 8 * half;
      p_lds[wave][row * 32 + lr]      = f2bf(s0[r]);
      p_lds[wave][row * 32 + 16 + lr] = f2bf(s1[r]);
    }
    // broadcast per-row alpha to per-column lanes (same-wave DS ops are in-order)
#pragma unroll
    for (int r = 0; r < 8; ++r)
      if (lr == r) red[wave][8 * half + r] = alpha[r];
    __builtin_amdgcn_wave_barrier();
    float acol = red[wave][lr];
    Frag bP = load_b_bf16(p_lds[wave] + lr * 32, 0, half);  // invariant across dt
    // t[d, n] = alpha*t[d, n] + g[d, m] * p[n, m]^T   (K = 32 columns)
#pragma unroll
    for (int dt = 0; dt < 8; ++dt) {
      Frag aG = load_a_bf16(gbase + (size_t)(dt * 16 + lr) * NVAL, m0, half);
#pragma unroll
      for (int r = 0; r < 8; ++r) acc[dt][r] *= acol;
      acc[dt] = wmma_bf16(aG, bP, acc[dt]);
    }
    __builtin_amdgcn_wave_barrier();
  }

  // final 1/l normalization, broadcast row stats to columns
#pragma unroll
  for (int r = 0; r < 8; ++r)
    if (lr == r) red[wave][8 * half + r] = lrun[r];
  __builtin_amdgcn_wave_barrier();
  float linv = 1.0f / red[wave][lr];

  unsigned short* orow = t_t + ((size_t)b * NVAL + n) * DIVAL;
#pragma unroll
  for (int dt = 0; dt < 8; ++dt) {
    Frag f;
#pragma unroll
    for (int r = 0; r < 8; ++r) f.u[r] = f2bf(acc[dt][r] * linv);
    *(uint4*)(orow + dt * 16 + half * 8) = f.q[0];
  }
}

// -------------------------------------------------------------------------
// Kernel 3: out 1x1 conv (DOxDI WMMA GEMM) + inference BN + residual, fp32 out
// -------------------------------------------------------------------------
__global__ void __launch_bounds__(128)
out_kernel(const unsigned short* __restrict__ t_t,
           const unsigned short* __restrict__ owb, const float* __restrict__ ob,
           const float* __restrict__ gma, const float* __restrict__ bta,
           const float* __restrict__ mu, const float* __restrict__ var,
           const float* __restrict__ x, float* __restrict__ out)
{
  const int b     = blockIdx.y;
  const int nblk  = blockIdx.x % (NVAL / 64);
  const int doblk = blockIdx.x / (NVAL / 64);
  const int tid   = threadIdx.x;
  const int wave  = tid >> 5, lane = tid & 31;
  const int half  = lane >> 4, lr = lane & 15;
  const int do0   = doblk * 16;
  const int n     = nblk * 64 + wave * 16 + lr;

  f32x8 acc = zero8();
  const unsigned short* trow = t_t + ((size_t)b * NVAL + n) * DIVAL;
  const unsigned short* wrow = owb + (size_t)(do0 + lr) * DIVAL;
#pragma unroll
  for (int kc = 0; kc < 4; ++kc) {
    Frag a  = load_a_bf16(wrow, kc * 32, half);
    Frag bb = load_b_bf16(trow, kc * 32, half);
    acc = wmma_bf16(a, bb, acc);
  }
#pragma unroll
  for (int r = 0; r < 8; ++r) {
    int d = do0 + 8 * half + r;
    float s  = gma[d] * __frsqrt_rn(var[d] + 1e-5f);
    float sh = (ob[d] - mu[d]) * s + bta[d];
    size_t idx = ((size_t)b * DOVAL + d) * NVAL + n;
    out[idx] = acc[r] * s + sh + x[idx];
  }
}

// -------------------------------------------------------------------------
extern "C" void kernel_launch(void* const* d_in, const int* in_sizes, int n_in,
                              void* d_out, int out_size, void* d_ws, size_t ws_size,
                              hipStream_t stream) {
  (void)in_sizes; (void)n_in; (void)out_size; (void)ws_size;
  const float* x   = (const float*)d_in[0];
  const float* tw  = (const float*)d_in[1];
  const float* tb  = (const float*)d_in[2];
  const float* pw  = (const float*)d_in[3];
  const float* pb  = (const float*)d_in[4];
  const float* gw  = (const float*)d_in[5];
  const float* gb  = (const float*)d_in[6];
  const float* ow  = (const float*)d_in[7];
  const float* ob  = (const float*)d_in[8];
  const float* gma = (const float*)d_in[9];
  const float* bta = (const float*)d_in[10];
  const float* mu  = (const float*)d_in[11];
  const float* var = (const float*)d_in[12];

  const size_t seg = (size_t)BVAL * NVAL * DIVAL;   // bf16 elements per tensor
  const size_t wseg = (size_t)DIVAL * CVAL;         // 32768 elements per weight
  unsigned short* theta_t = (unsigned short*)d_ws;  // [B][N][DI]
  unsigned short* phi_t   = theta_t + seg;          // [B][N][DI]
  unsigned short* g_dm    = phi_t + seg;            // [B][DI][N]
  unsigned short* t_t     = g_dm + seg;             // [B][N][DI]
  unsigned short* twb     = t_t + seg;              // [DI][C] bf16
  unsigned short* pwb     = twb + wseg;
  unsigned short* gwb     = pwb + wseg;
  unsigned short* owb     = gwb + wseg;             // [DO][DI] bf16

  dim3 blk(128);
  const int wcnt = (int)wseg;                        // DI*C == DO*DI == 32768
  cvt_kernel<<<dim3((wcnt + 255) / 256), dim3(256), 0, stream>>>(tw, twb, wcnt);
  cvt_kernel<<<dim3((wcnt + 255) / 256), dim3(256), 0, stream>>>(pw, pwb, wcnt);
  cvt_kernel<<<dim3((wcnt + 255) / 256), dim3(256), 0, stream>>>(gw, gwb, wcnt);
  cvt_kernel<<<dim3((wcnt + 255) / 256), dim3(256), 0, stream>>>(ow, owb, wcnt);

  // 8 KB dynamic LDS regardless of which staging path the device build chose.
  proj_kernel<<<dim3((NVAL / 64) * (DIVAL / 16), BVAL), blk, 8192, stream>>>(
      x, twb, tb, pwb, pb, gwb, gb, theta_t, phi_t, g_dm);
  attn_kernel<<<dim3(NVAL / 64, BVAL), blk, 0, stream>>>(theta_t, phi_t, g_dm, t_t);
  out_kernel<<<dim3((NVAL / 64) * (DOVAL / 16), BVAL), blk, 0, stream>>>(
      t_t, owb, ob, gma, bta, mu, var, x, (float*)d_out);
}
